// TransformerBlock_66571993088137
// MI455X (gfx1250) — compile-verified, matrix operand order checked by reading, not run on hardware
//
#include <hip/hip_runtime.h>
#include <hip/hip_bf16.h>

typedef __bf16 bf16;
typedef __attribute__((ext_vector_type(16))) __bf16 v16bf;
typedef __attribute__((ext_vector_type(8)))  float  v8f;
typedef int vint4 __attribute__((vector_size(16)));

#define WMMA_BF16(a, b, c) \
    __builtin_amdgcn_wmma_f32_16x16x32_bf16(false, (a), false, (b), (short)0, (c), false, false)

// Async global->LDS path (ASYNCcnt / s_wait_asynccnt), gated so both device
// passes with and without the builtins (and the host pass) compile clean.
#if __has_builtin(__builtin_amdgcn_global_load_async_to_lds_b128) && \
    __has_builtin(__builtin_amdgcn_s_wait_asynccnt)
#define USE_ASYNC_LDS 1
#else
#define USE_ASYNC_LDS 0
#endif

#if USE_ASYNC_LDS
__device__ __forceinline__ void async_copy_b128(const void* g, void* l) {
    // Builtin takes generic int4* pointers (per toolchain diagnostic); the
    // backend inserts the flat->global / flat->LDS address-space casts.
    __builtin_amdgcn_global_load_async_to_lds_b128((vint4*)g, (vint4*)l, 0, 0);
}
#endif

// ---------------------------------------------------------------------------
// Fragment loader: 16-bit A/B operand, wave32 layout (CDNA5 ISA).
// Lane l holds row (row0 + (l&15)); koff=(l>=16?8:0); VGPR0..3 <- K koff..koff+7,
// VGPR4..7 <- K koff+16..koff+23.  With a 16B-aligned LDS row stride the two
// 8-elem runs merge into 2x ds_load_b128 (verified in round-1/2 disasm).
// ---------------------------------------------------------------------------
__device__ __forceinline__ v16bf load_frag(const bf16* __restrict__ base, int ld,
                                           int row0, int k0, int lane) {
    union { v16bf v; uint2 d2[4]; } f;
    const bf16* p = base + (size_t)(row0 + (lane & 15)) * ld + k0 + ((lane >> 4) << 3);
    f.d2[0] = *(const uint2*)(p + 0);
    f.d2[1] = *(const uint2*)(p + 4);
    f.d2[2] = *(const uint2*)(p + 16);
    f.d2[3] = *(const uint2*)(p + 20);
    return f.v;
}

// ---------------------------------------------------------------------------
// Transpose + convert: W (R x C, f32, row-major) -> Wt (C x R, bf16)
// ---------------------------------------------------------------------------
__global__ __launch_bounds__(256) void transpose_to_bf16_kernel(
    const float* __restrict__ W, bf16* __restrict__ Wt, int R, int C) {
    __shared__ float tile[32][33];
    const int bc = blockIdx.x * 32, br = blockIdx.y * 32;
    const int tx = threadIdx.x & 31, ty = threadIdx.x >> 5;
    #pragma unroll
    for (int i = 0; i < 32; i += 8)
        tile[ty + i][tx] = W[(size_t)(br + ty + i) * C + bc + tx];
    __syncthreads();
    #pragma unroll
    for (int i = 0; i < 32; i += 8)
        Wt[(size_t)(bc + ty + i) * R + br + tx] = (bf16)tile[tx][ty + i];
}

// ---------------------------------------------------------------------------
// RMSNorm: one block per row. y = x * rsqrt(EPS + mean(x^2)) * gamma  (bf16 out)
// ---------------------------------------------------------------------------
__global__ __launch_bounds__(256) void rmsnorm_kernel(
    const float* __restrict__ X, const float* __restrict__ gamma,
    bf16* __restrict__ Y, int D) {
    const int row = blockIdx.x;
    const float* x = X + (size_t)row * D;
    float ss = 0.f;
    for (int i = threadIdx.x; i < D; i += 256) { float v = x[i]; ss += v * v; }
    #pragma unroll
    for (int off = 16; off > 0; off >>= 1) ss += __shfl_xor(ss, off);
    __shared__ float red[8];
    if ((threadIdx.x & 31) == 0) red[threadIdx.x >> 5] = ss;
    __syncthreads();
    float tot = 0.f;
    #pragma unroll
    for (int w = 0; w < 8; w++) tot += red[w];
    const float inv = rsqrtf(1e-7f + tot / (float)D);
    bf16* y = Y + (size_t)row * D;
    for (int i = threadIdx.x; i < D; i += 256)
        y[i] = (bf16)(x[i] * inv * gamma[i]);
}

// ---------------------------------------------------------------------------
// bf16 WMMA GEMM:  C[MxN] = A[MxK] * Bt[NxK]^T  (+bias, +residual)
// Block tile 128x128, 8 waves of 64x32 (4x2 wmma tiles), K-step 64.
// Staging: async global->LDS double buffer (ASYNCcnt) when available,
// else register double buffer.  Transfers for tile t+1 overlap the 16 WMMAs
// on tile t.
// ---------------------------------------------------------------------------
#define LDT 80  // padded K-stride (64 + 16); 160 B row = 16B-aligned for b128

template <bool BIAS, bool RES, bool TOBF>
__global__ __launch_bounds__(256) void gemm_bf16_kernel(
    const bf16* __restrict__ A, const bf16* __restrict__ Bt,
    int M, int N, int K,
    const float* __restrict__ bias, const float* __restrict__ residual,
    float* __restrict__ Cf, bf16* __restrict__ Cb) {
#if USE_ASYNC_LDS
    __shared__ bf16 lA[2][128 * LDT];
    __shared__ bf16 lB[2][128 * LDT];
#else
    __shared__ bf16 lA[1][128 * LDT];
    __shared__ bf16 lB[1][128 * LDT];
#endif

    const int tid  = threadIdx.x;
    const int lane = tid & 31;
    const int wave = tid >> 5;
    const int bm = blockIdx.y * 128;
    const int bn = blockIdx.x * 128;
    const int wm = (wave & 1) * 64;   // wave row base in block tile
    const int wn = (wave >> 1) * 32;  // wave col base in block tile

    v8f acc[4][2];
    #pragma unroll
    for (int i = 0; i < 4; i++)
        #pragma unroll
        for (int j = 0; j < 2; j++) acc[i][j] = v8f{};

    // 16 WMMAs on one staged 128x64 tile pair
    auto compute = [&](const bf16* la, const bf16* lb) {
        #pragma unroll
        for (int kc = 0; kc < 2; kc++) {
            v16bf b0 = load_frag(lb, LDT, wn +  0, kc * 32, lane);
            v16bf b1 = load_frag(lb, LDT, wn + 16, kc * 32, lane);
            #pragma unroll
            for (int i = 0; i < 4; i++) {
                v16bf a = load_frag(la, LDT, wm + i * 16, kc * 32, lane);
                acc[i][0] = WMMA_BF16(a, b0, acc[i][0]);
                acc[i][1] = WMMA_BF16(a, b1, acc[i][1]);
            }
        }
    };

#if USE_ASYNC_LDS
    // 128 rows x 64 bf16 per tile = 1024 b128 slots; 4 async copies per thread.
    auto issueTile = [&](const bf16* __restrict__ base, int rbase, int k0, bf16* lds) {
        #pragma unroll
        for (int i = 0; i < 4; i++) {
            const int s = tid + i * 256;
            const int r = s >> 3, c = (s & 7) * 8;
            async_copy_b128(&base[(size_t)(rbase + r) * K + k0 + c], &lds[r * LDT + c]);
        }
    };
    issueTile(A, bm, 0, lA[0]);
    issueTile(Bt, bn, 0, lB[0]);
    __builtin_amdgcn_s_wait_asynccnt(0);
    __syncthreads();
    int cur = 0;
    for (int k0 = 0; k0 < K; k0 += 64) {
        const bool more = (k0 + 64) < K;
        if (more) {                          // DMA next tile while we compute
            issueTile(A, bm, k0 + 64, lA[cur ^ 1]);
            issueTile(Bt, bn, k0 + 64, lB[cur ^ 1]);
        }
        compute(lA[cur], lB[cur]);
        if (more) __builtin_amdgcn_s_wait_asynccnt(0);
        __syncthreads();
        cur ^= 1;
    }
#else
    // Register double buffer fallback.
    uint4 stA[4], stB[4];
    auto loadG = [&](const bf16* __restrict__ base, int rbase, int k0, uint4* st) {
        #pragma unroll
        for (int i = 0; i < 4; i++) {
            const int s = tid + i * 256;
            const int r = s >> 3, c = (s & 7) * 8;
            st[i] = *(const uint4*)&base[(size_t)(rbase + r) * K + k0 + c];
        }
    };
    auto storeL = [&](bf16* lds, const uint4* st) {
        #pragma unroll
        for (int i = 0; i < 4; i++) {
            const int s = tid + i * 256;
            const int r = s >> 3, c = (s & 7) * 8;
            *(uint4*)&lds[r * LDT + c] = st[i];
        }
    };
    loadG(A, bm, 0, stA);
    loadG(Bt, bn, 0, stB);
    storeL(lA[0], stA);
    storeL(lB[0], stB);
    __syncthreads();
    for (int k0 = 0; k0 < K; k0 += 64) {
        const bool more = (k0 + 64) < K;
        if (more) {
            loadG(A, bm, k0 + 64, stA);
            loadG(Bt, bn, k0 + 64, stB);
        }
        compute(lA[0], lB[0]);
        __syncthreads();
        if (more) {
            storeL(lA[0], stA);
            storeL(lB[0], stB);
        }
        __syncthreads();
    }
#endif

    const int cn = lane & 15;
    const int rb = (lane >> 4) * 8;
    #pragma unroll
    for (int j = 0; j < 2; j++) {
        const int gn = bn + wn + j * 16 + cn;
        float bv = 0.f;
        if constexpr (BIAS) bv = bias[gn];
        #pragma unroll
        for (int i = 0; i < 4; i++) {
            #pragma unroll
            for (int r = 0; r < 8; r++) {
                const int gm = bm + wm + i * 16 + rb + r;
                const size_t idx = (size_t)gm * N + gn;
                float v = acc[i][j][r] + bv;
                if constexpr (RES) v += residual[idx];
                if constexpr (TOBF) Cb[idx] = (bf16)v;
                else               Cf[idx] = v;
            }
        }
    }
}

// ---------------------------------------------------------------------------
// RoPE: q,k (S x D f32) -> bf16, half-rotation per 128-wide head.
// ---------------------------------------------------------------------------
__global__ void rope_kernel(const float* __restrict__ Qf, const float* __restrict__ Kf,
                            bf16* __restrict__ Qr, bf16* __restrict__ Kr, int S, int D) {
    const int idx = blockIdx.x * blockDim.x + threadIdx.x;  // over S * D/2
    if (idx >= S * (D / 2)) return;
    const int s = idx / (D / 2);
    const int rem = idx - s * (D / 2);
    const int h = rem >> 6;        // head (64 pairs per head)
    const int d = rem & 63;        // pair index within head
    const int c1 = h * 128 + d, c2 = c1 + 64;
    const float ang = (float)s * __powf(10000.0f, -(float)d * (1.0f / 64.0f));
    float cs, sn;
    __sincosf(ang, &sn, &cs);
    const size_t r = (size_t)s * D;
    const float q1 = Qf[r + c1], q2 = Qf[r + c2];
    const float k1 = Kf[r + c1], k2 = Kf[r + c2];
    Qr[r + c1] = (bf16)(q1 * cs - q2 * sn);
    Qr[r + c2] = (bf16)(q2 * cs + q1 * sn);
    Kr[r + c1] = (bf16)(k1 * cs - k2 * sn);
    Kr[r + c2] = (bf16)(k2 * cs + k1 * sn);
}

// ---------------------------------------------------------------------------
// Flash attention: block = (128 q-rows, one head). 8 waves x 16 q-rows.
// Online softmax in fp32 registers; QK^T and PV via bf16 WMMA.
// LDS: Q(128x136) + K(128x136, reused as P stage) + Vt(128x136) = ~102 KB.
// ---------------------------------------------------------------------------
#define LDH 136

__global__ __launch_bounds__(256) void attn_kernel(
    const bf16* __restrict__ Q, const bf16* __restrict__ Kb, const bf16* __restrict__ V,
    bf16* __restrict__ O, int S, int D) {
    __shared__ bf16 lQ[128 * LDH];
    __shared__ bf16 lK[128 * LDH];   // reused as P staging after scores
    __shared__ bf16 lVt[128 * LDH];

    const int tid = threadIdx.x, lane = tid & 31, wave = tid >> 5;
    const int h = blockIdx.y;
    const int q0 = blockIdx.x * 128;
    const int colBase = h * 128;
    const int wq = wave * 16;
    const int cn = lane & 15;
    const int rOff = (lane >> 4) * 8;

    for (int s = tid; s < 4096; s += 256) {
        int r = s >> 5, c = (s & 31) * 4;
        *(uint2*)&lQ[r * LDH + c] = *(const uint2*)&Q[(size_t)(q0 + r) * D + colBase + c];
    }

    v8f o_acc[8];
    #pragma unroll
    for (int i = 0; i < 8; i++) o_acc[i] = v8f{};
    float rm[8], rl[8];
    #pragma unroll
    for (int r = 0; r < 8; r++) { rm[r] = -__builtin_inff(); rl[r] = 0.f; }

    const float scale = 0.08838834764831845f;  // 1/sqrt(128)

    for (int kb = 0; kb <= blockIdx.x; kb++) {
        const int k0 = kb * 128;
        __syncthreads();  // prev iteration's P@V (and initial lQ stores) done
        for (int s = tid; s < 4096; s += 256) {
            int r = s >> 5, c = (s & 31) * 4;
            *(uint2*)&lK[r * LDH + c] = *(const uint2*)&Kb[(size_t)(k0 + r) * D + colBase + c];
        }
        for (int s = tid; s < 4096; s += 256) {
            int kr = s >> 5, c = (s & 31) * 4;
            uint2 raw = *(const uint2*)&V[(size_t)(k0 + kr) * D + colBase + c];
            bf16 tmp[4];
            *(uint2*)tmp = raw;
            lVt[(c + 0) * LDH + kr] = tmp[0];
            lVt[(c + 1) * LDH + kr] = tmp[1];
            lVt[(c + 2) * LDH + kr] = tmp[2];
            lVt[(c + 3) * LDH + kr] = tmp[3];
        }
        // pull next K/V block toward L2 while we compute (global_prefetch_b8)
        if (kb < blockIdx.x) {
            const size_t nrow = (size_t)(k0 + 128 + (tid >> 1)) * D + colBase + (tid & 1) * 64;
            __builtin_prefetch(&Kb[nrow], 0, 3);
            __builtin_prefetch(&V[nrow], 0, 3);
        }
        __syncthreads();

        // scores: 8 key tiles of 16, K-dim = hd = 128 (4 chunks of 32)
        v8f sc[8];
        #pragma unroll
        for (int nt = 0; nt < 8; nt++) {
            v8f a = v8f{};
            #pragma unroll
            for (int kc = 0; kc < 4; kc++) {
                v16bf qa = load_frag(lQ, LDH, wq, kc * 32, lane);
                v16bf kf = load_frag(lK, LDH, nt * 16, kc * 32, lane);
                a = WMMA_BF16(qa, kf, a);
            }
            sc[nt] = a;
        }

        // causal mask + scale + row max
        float mx[8];
        #pragma unroll
        for (int r = 0; r < 8; r++) mx[r] = -__builtin_inff();
        #pragma unroll
        for (int nt = 0; nt < 8; nt++) {
            const int kg = k0 + nt * 16 + cn;
            #pragma unroll
            for (int r = 0; r < 8; r++) {
                const int qg = q0 + wq + rOff + r;
                float v = sc[nt][r] * scale;
                v = (kg <= qg) ? v : -__builtin_inff();
                sc[nt][r] = v;
                mx[r] = fmaxf(mx[r], v);
            }
        }
        #pragma unroll
        for (int r = 0; r < 8; r++) {
            float m = mx[r];
            m = fmaxf(m, __shfl_xor(m, 1));
            m = fmaxf(m, __shfl_xor(m, 2));
            m = fmaxf(m, __shfl_xor(m, 4));
            m = fmaxf(m, __shfl_xor(m, 8));
            mx[r] = m;
        }
        float sc_old[8];
        #pragma unroll
        for (int r = 0; r < 8; r++) {
            const float mnew = fmaxf(rm[r], mx[r]);
            sc_old[r] = __expf(rm[r] - mnew);
            rm[r] = mnew;
        }
        float rs[8];
        #pragma unroll
        for (int r = 0; r < 8; r++) rs[r] = 0.f;
        #pragma unroll
        for (int nt = 0; nt < 8; nt++)
            #pragma unroll
            for (int r = 0; r < 8; r++) {
                const float p = __expf(sc[nt][r] - rm[r]);
                sc[nt][r] = p;
                rs[r] += p;
            }
        #pragma unroll
        for (int r = 0; r < 8; r++) {
            float t = rs[r];
            t += __shfl_xor(t, 1);
            t += __shfl_xor(t, 2);
            t += __shfl_xor(t, 4);
            t += __shfl_xor(t, 8);
            rl[r] = rl[r] * sc_old[r] + t;
        }
        #pragma unroll
        for (int dt = 0; dt < 8; dt++)
            #pragma unroll
            for (int r = 0; r < 8; r++) o_acc[dt][r] *= sc_old[r];

        __syncthreads();  // all waves done reading lK -> reuse as P stage
        bf16* lpw = &lK[wq * LDH];  // wave-private 16 x 128 region
        #pragma unroll
        for (int nt = 0; nt < 8; nt++)
            #pragma unroll
            for (int r = 0; r < 8; r++)
                lpw[(rOff + r) * LDH + nt * 16 + cn] = (bf16)sc[nt][r];

        // O += P @ V : 8 hd-tiles of 16, K-dim = 128 keys (4 chunks of 32)
        #pragma unroll
        for (int dt = 0; dt < 8; dt++) {
            #pragma unroll
            for (int kc = 0; kc < 4; kc++) {
                v16bf pa = load_frag(lpw, LDH, 0, kc * 32, lane);
                v16bf vb = load_frag(lVt, LDH, dt * 16, kc * 32, lane);
                o_acc[dt] = WMMA_BF16(pa, vb, o_acc[dt]);
            }
        }
    }

    #pragma unroll
    for (int dt = 0; dt < 8; dt++)
        #pragma unroll
        for (int r = 0; r < 8; r++) {
            const int qg = q0 + wq + rOff + r;
            O[(size_t)qg * D + colBase + dt * 16 + cn] = (bf16)(o_acc[dt][r] / rl[r]);
        }
}

// ---------------------------------------------------------------------------
// SwiGLU gate: g = silu(x1) * x2  (bf16 in/out, fp32 math)
// ---------------------------------------------------------------------------
__global__ void silu_mul_kernel(const bf16* __restrict__ x1, const bf16* __restrict__ x2,
                                bf16* __restrict__ g, size_t n) {
    size_t i = blockIdx.x * (size_t)blockDim.x + threadIdx.x;
    if (i < n) {
        const float a = (float)x1[i], b = (float)x2[i];
        g[i] = (bf16)(a / (1.f + __expf(-a)) * b);
    }
}

// ---------------------------------------------------------------------------
// Host launcher
// ---------------------------------------------------------------------------
extern "C" void kernel_launch(void* const* d_in, const int* in_sizes, int n_in,
                              void* d_out, int out_size, void* d_ws, size_t ws_size,
                              hipStream_t stream) {
    const int S = 2048, D = 4096, Hd = 11008, H = 32;
    (void)in_sizes; (void)n_in; (void)out_size; (void)ws_size;

    const float* X      = (const float*)d_in[0];
    const float* Wq     = (const float*)d_in[1];
    const float* Wk     = (const float*)d_in[2];
    const float* Wv     = (const float*)d_in[3];
    const float* Wo     = (const float*)d_in[4];
    const float* bo     = (const float*)d_in[5];
    const float* gamma1 = (const float*)d_in[6];
    const float* gamma2 = (const float*)d_in[7];
    const float* W1     = (const float*)d_in[8];
    const float* W2     = (const float*)d_in[9];
    const float* W3     = (const float*)d_in[10];
    float* out = (float*)d_out;

    char* w = (char*)d_ws;
    auto alloc = [&](size_t bytes) -> char* {
        char* p = w;
        w += (bytes + 255) & ~(size_t)255;
        return p;
    };
    bf16* WqT  = (bf16*)alloc((size_t)D * D * 2);
    bf16* WkT  = (bf16*)alloc((size_t)D * D * 2);
    bf16* WvT  = (bf16*)alloc((size_t)D * D * 2);
    bf16* WoT  = (bf16*)alloc((size_t)D * D * 2);
    bf16* W1T  = (bf16*)alloc((size_t)Hd * D * 2);
    bf16* W2T  = (bf16*)alloc((size_t)Hd * D * 2);
    bf16* W3T  = (bf16*)alloc((size_t)D * Hd * 2);
    bf16* hb   = (bf16*)alloc((size_t)S * D * 2);
    float* qf  = (float*)alloc((size_t)S * D * 4);
    float* kf  = (float*)alloc((size_t)S * D * 4);
    bf16* qr   = (bf16*)alloc((size_t)S * D * 2);
    bf16* kr   = (bf16*)alloc((size_t)S * D * 2);
    bf16* vb   = (bf16*)alloc((size_t)S * D * 2);
    bf16* attn = (bf16*)alloc((size_t)S * D * 2);
    float* xmid = (float*)alloc((size_t)S * D * 4);
    bf16* h2   = (bf16*)alloc((size_t)S * D * 2);
    bf16* x1b  = (bf16*)alloc((size_t)S * Hd * 2);
    bf16* x2b  = (bf16*)alloc((size_t)S * Hd * 2);
    bf16* gb   = (bf16*)alloc((size_t)S * Hd * 2);

    // 1. Weight transpose + bf16 convert
    transpose_to_bf16_kernel<<<dim3(D / 32, D / 32), 256, 0, stream>>>(Wq, WqT, D, D);
    transpose_to_bf16_kernel<<<dim3(D / 32, D / 32), 256, 0, stream>>>(Wk, WkT, D, D);
    transpose_to_bf16_kernel<<<dim3(D / 32, D / 32), 256, 0, stream>>>(Wv, WvT, D, D);
    transpose_to_bf16_kernel<<<dim3(D / 32, D / 32), 256, 0, stream>>>(Wo, WoT, D, D);
    transpose_to_bf16_kernel<<<dim3(Hd / 32, D / 32), 256, 0, stream>>>(W1, W1T, D, Hd);
    transpose_to_bf16_kernel<<<dim3(Hd / 32, D / 32), 256, 0, stream>>>(W2, W2T, D, Hd);
    transpose_to_bf16_kernel<<<dim3(D / 32, Hd / 32), 256, 0, stream>>>(W3, W3T, Hd, D);

    // 2. RMSNorm 1
    rmsnorm_kernel<<<S, 256, 0, stream>>>(X, gamma1, hb, D);

    // 3. QKV projections (q,k fp32 for RoPE; v straight to bf16)
    dim3 gDD(D / 128, S / 128);
    gemm_bf16_kernel<false, false, false><<<gDD, 256, 0, stream>>>(
        hb, WqT, S, D, D, nullptr, nullptr, qf, nullptr);
    gemm_bf16_kernel<false, false, false><<<gDD, 256, 0, stream>>>(
        hb, WkT, S, D, D, nullptr, nullptr, kf, nullptr);
    gemm_bf16_kernel<false, false, true><<<gDD, 256, 0, stream>>>(
        hb, WvT, S, D, D, nullptr, nullptr, nullptr, vb);

    // 4. RoPE
    {
        const int n = S * (D / 2);
        rope_kernel<<<(n + 255) / 256, 256, 0, stream>>>(qf, kf, qr, kr, S, D);
    }

    // 5. Flash attention
    attn_kernel<<<dim3(S / 128, H), 256, 0, stream>>>(qr, kr, vb, attn, S, D);

    // 6. Output projection + bias + residual -> xmid (fp32)
    gemm_bf16_kernel<true, true, false><<<gDD, 256, 0, stream>>>(
        attn, WoT, S, D, D, bo, X, xmid, nullptr);

    // 7. RMSNorm 2
    rmsnorm_kernel<<<S, 256, 0, stream>>>(xmid, gamma2, h2, D);

    // 8. MLP up projections
    dim3 gDH(Hd / 128, S / 128);
    gemm_bf16_kernel<false, false, true><<<gDH, 256, 0, stream>>>(
        h2, W1T, S, Hd, D, nullptr, nullptr, nullptr, x1b);
    gemm_bf16_kernel<false, false, true><<<gDH, 256, 0, stream>>>(
        h2, W2T, S, Hd, D, nullptr, nullptr, nullptr, x2b);

    // 9. SwiGLU gate
    {
        const size_t n = (size_t)S * Hd;
        silu_mul_kernel<<<(unsigned)((n + 255) / 256), 256, 0, stream>>>(x1b, x2b, gb, n);
    }

    // 10. Down projection + residual -> d_out (fp32)
    gemm_bf16_kernel<false, true, false><<<gDD, 256, 0, stream>>>(
        gb, W3T, S, D, Hd, nullptr, xmid, out, nullptr);
}
